// Net_21904333209612
// MI455X (gfx1250) — matrix operand order repaired and checked by reading, not compiled, not run on hardware
//
#include <hip/hip_runtime.h>
#include <math.h>
#include <stdint.h>

typedef __attribute__((ext_vector_type(16))) _Float16 v16h;
typedef __attribute__((ext_vector_type(8)))  _Float16 h8;
typedef __attribute__((ext_vector_type(8)))  float    v8f;
typedef __attribute__((ext_vector_type(4)))  float    f32x4;

#define BATCH 512
#define TSTEP 5
#define NPROB (BATCH * TSTEP)   // 2560
#define NEq 17
#define NVa 24
#define AROW 25                 // LDS row stride (gcd(25,64)=1 -> conflict-free columns)
#define PDHG_ITERS 400
#define POW_ITERS 25

union HV { v16h v; _Float16 h[16]; };
union FV { v8f  v; float    f[8];  };

// ---------------------------------------------------------------------------
// conv1 (1->10, 11x11 valid on 50x50 -> 40x40) + maxpool2 + relu -> (B,10,20,20)
// ---------------------------------------------------------------------------
__global__ void __launch_bounds__(256)
conv1_pool(const float* __restrict__ img, const float* __restrict__ w,
           const float* __restrict__ bias, float* __restrict__ out)
{
    int idx = blockIdx.x * blockDim.x + threadIdx.x;
    if (idx >= BATCH * 10 * 20 * 20) return;
    int x = idx % 20;
    int y = (idx / 20) % 20;
    int c = (idx / 400) % 10;
    int b = idx / 4000;
    const float* im = img + (size_t)b * 2500;
    const float* wc = w + c * 121;
    float bs = bias[c];
    float m = -1e30f;
    for (int dy = 0; dy < 2; ++dy) {
        for (int dx = 0; dx < 2; ++dx) {
            int Y = 2 * y + dy, X = 2 * x + dx;
            float s = bs;
            for (int ky = 0; ky < 11; ++ky) {
#pragma unroll
                for (int kx = 0; kx < 11; ++kx)
                    s += im[(Y + ky) * 50 + (X + kx)] * wc[ky * 11 + kx];
            }
            m = fmaxf(m, s);
        }
    }
    out[idx] = fmaxf(m, 0.f);
}

// ---------------------------------------------------------------------------
// conv2 (10->20, 5x5 valid on 20x20 -> 16x16) + maxpool4 + relu -> flat (B,320)
// ---------------------------------------------------------------------------
__global__ void __launch_bounds__(256)
conv2_pool(const float* __restrict__ in, const float* __restrict__ w,
           const float* __restrict__ bias, float* __restrict__ out)
{
    int idx = blockIdx.x * blockDim.x + threadIdx.x;
    if (idx >= BATCH * 20 * 4 * 4) return;
    int x = idx % 4;
    int y = (idx / 4) % 4;
    int c = (idx / 16) % 20;
    int b = idx / 320;
    float bs = bias[c];
    float m = -1e30f;
    for (int dy = 0; dy < 4; ++dy) {
        for (int dx = 0; dx < 4; ++dx) {
            int Y = 4 * y + dy, X = 4 * x + dx;
            float s = bs;
            for (int ic = 0; ic < 10; ++ic) {
                const float* ip = in + (size_t)b * 4000 + ic * 400;
                const float* wp = w + c * 250 + ic * 25;
#pragma unroll
                for (int ky = 0; ky < 5; ++ky)
#pragma unroll
                    for (int kx = 0; kx < 5; ++kx)
                        s += ip[(Y + ky) * 20 + (X + kx)] * wp[ky * 5 + kx];
            }
            m = fmaxf(m, s);
        }
    }
    out[(size_t)b * 320 + c * 16 + y * 4 + x] = fmaxf(m, 0.f);
}

// ---------------------------------------------------------------------------
// Weight packing: W(N,K) f32 -> Wp(Nrun,Kpad) f16, zero padded.
// ---------------------------------------------------------------------------
__global__ void __launch_bounds__(256)
pack_w(const float* __restrict__ W, _Float16* __restrict__ Wp,
       int N, int K, int Nrun, int Kpad)
{
    int idx = blockIdx.x * blockDim.x + threadIdx.x;
    if (idx >= Nrun * Kpad) return;
    int n = idx / Kpad, k = idx - n * Kpad;
    float v = (n < N && k < K) ? W[(size_t)n * K + k] : 0.f;
    Wp[idx] = (_Float16)v;
}

__global__ void __launch_bounds__(256)
pack_b(const float* __restrict__ B, float* __restrict__ Bp, int N, int Nrun)
{
    int idx = blockIdx.x * blockDim.x + threadIdx.x;
    if (idx >= Nrun) return;
    Bp[idx] = (idx < N) ? B[idx] : 0.f;
}

// ---------------------------------------------------------------------------
// WMMA GEMM, fully padded (no bounds checks anywhere):
//   C[M, Nrun] = act( X[M, Kpad](f32) @ Wp[Nrun, Kpad](f16)^T + bp[Nrun] )
// M % 16 == 0, Nrun % 16 == 0, Kpad % 32 == 0. One 16x16 tile per wave.
// ---------------------------------------------------------------------------
__global__ void __launch_bounds__(128)
wmma_gemm(const float* __restrict__ X, const _Float16* __restrict__ Wp,
          const float* __restrict__ bp, float* __restrict__ C,
          int M, int Nrun, int Kpad, int relu)
{
    const int wave = (blockIdx.x * blockDim.x + threadIdx.x) >> 5;
    const int lane = threadIdx.x & 31;
    const int ntiles = Nrun >> 4;
    const int mt = wave / ntiles;
    const int nt = wave - mt * ntiles;
    if (mt >= (M >> 4)) return;          // wave-uniform exit

    const int lh  = lane >> 4;           // lane half
    const int l16 = lane & 15;
    const int m = mt * 16 + l16;         // A row owned by this lane
    const int n = nt * 16 + l16;         // B column owned by this lane

    FV acc;
#pragma unroll
    for (int r = 0; r < 8; ++r) acc.f[r] = 0.f;

    // A f16 16x32: elem i -> k = ((i<8)? i : i+8) + 8*lh  (two 8-float runs)
    // B f16 32x16: elem i -> k = ((i<8)? i : i+8) + 8*lh  (two 8-half runs)
    const float*    xrow = X  + (size_t)m * Kpad + lh * 8;
    const _Float16* wrow = Wp + (size_t)n * Kpad + lh * 8;

    for (int k0 = 0; k0 < Kpad; k0 += 32) {
        const f32x4* xp = (const f32x4*)(xrow + k0);
        f32x4 x0 = xp[0];   // k +0..3
        f32x4 x1 = xp[1];   // k +4..7
        f32x4 x2 = xp[4];   // k +16..19
        f32x4 x3 = xp[5];   // k +20..23
        const h8* wp8 = (const h8*)(wrow + k0);
        h8 w0 = wp8[0];
        h8 w1 = wp8[2];

        HV a, bm;
#pragma unroll
        for (int i = 0; i < 4; ++i) {
            a.h[i]      = (_Float16)x0[i];
            a.h[4 + i]  = (_Float16)x1[i];
            a.h[8 + i]  = (_Float16)x2[i];
            a.h[12 + i] = (_Float16)x3[i];
        }
#pragma unroll
        for (int i = 0; i < 8; ++i) {
            bm.h[i]     = w0[i];
            bm.h[8 + i] = w1[i];
        }
        acc.v = __builtin_amdgcn_wmma_f32_16x16x32_f16(
            false, a.v, false, bm.v, (short)0, acc.v, false, false);
    }

    const int no = nt * 16 + l16;
    const float bn = bp[no];
    float* crow = C + (size_t)(mt * 16 + 8 * lh) * Nrun + no;
#pragma unroll
    for (int r = 0; r < 8; ++r) {
        float v = acc.f[r] + bn;
        if (relu) v = fmaxf(v, 0.f);
        crow[(size_t)r * Nrun] = v;
    }
}

// ---------------------------------------------------------------------------
// h = concat(e (B,100 @stride112), xtraj (B,45)) -> (B,160), pad cols zeroed
// ---------------------------------------------------------------------------
__global__ void __launch_bounds__(256)
concat_h(const float* __restrict__ e, const float* __restrict__ xtraj,
         float* __restrict__ h)
{
    int idx = blockIdx.x * blockDim.x + threadIdx.x;
    if (idx >= BATCH * 160) return;
    int b = idx / 160, j = idx - b * 160;
    float v;
    if (j < 100)      v = e[(size_t)b * 112 + j];
    else if (j < 145) v = xtraj[(size_t)b * 45 + (j - 100)];
    else              v = 0.f;
    h[idx] = v;
}

// ---------------------------------------------------------------------------
// Build A (17x24) and b (17) per (sample,t) problem.
// prnet stride 32, vnet stride 48 (padded GEMM outputs).
// ---------------------------------------------------------------------------
__global__ void __launch_bounds__(256)
build_Ab(const float* __restrict__ xtraj, const float* __restrict__ x,
         const float* __restrict__ vnet, const float* __restrict__ prnet,
         float* __restrict__ A, float* __restrict__ bvec)
{
    int p = blockIdx.x * blockDim.x + threadIdx.x;
    if (p >= NPROB) return;
    int b = p / TSTEP, t = p - b * TSTEP;

    float r0   = xtraj[(size_t)b * 45 + 0 + t];
    float r1   = xtraj[(size_t)b * 45 + 5 + t];
    float ddr0 = xtraj[(size_t)b * 45 + 30 + t];
    float ddr1 = xtraj[(size_t)b * 45 + 35 + t];
    float ddr2 = xtraj[(size_t)b * 45 + 40 + t];
    float fc[8], pe[4], fce[8], vv[8], pr[4];
#pragma unroll
    for (int j = 0; j < 8; ++j) fc[j]  = x[(size_t)b * 160 + 20 + 5 * j + t];
#pragma unroll
    for (int j = 0; j < 4; ++j) pe[j]  = x[(size_t)b * 160 + 60 + 5 * j + t];
#pragma unroll
    for (int j = 0; j < 8; ++j) fce[j] = x[(size_t)b * 160 + 80 + 5 * j + t];
#pragma unroll
    for (int j = 0; j < 8; ++j) vv[j]  = vnet[(size_t)b * 48 + 5 * j + t];
#pragma unroll
    for (int j = 0; j < 4; ++j) pr[j]  = prnet[(size_t)b * 32 + 5 * j + t];

    float* Ap = A + (size_t)p * (NEq * NVa);
    for (int i = 0; i < NEq * NVa; ++i) Ap[i] = 0.f;
    float bl[NEq];
#pragma unroll
    for (int i = 0; i < NEq; ++i) bl[i] = 0.f;

    Ap[0*NVa+4]=1.f; Ap[0*NVa+5]=1.f; Ap[0*NVa+6]=1.f; Ap[0*NVa+7]=1.f;
    Ap[0*NVa+8]=1.f; Ap[0*NVa+9]=1.f;                       bl[0]=ddr0;
    Ap[1*NVa+10]=1.f; Ap[1*NVa+11]=1.f;                     bl[1]=ddr1;
    Ap[2*NVa+6]=pr[0]-r0;  Ap[2*NVa+4]=-(pr[2]-r1);
    Ap[2*NVa+7]=pr[1]-r0;  Ap[2*NVa+5]=-(pr[3]-r1);
    Ap[2*NVa+10]=pe[0]-r0; Ap[2*NVa+8]=-(pe[2]-r1);
    Ap[2*NVa+11]=pe[1]-r0; Ap[2*NVa+9]=-(pe[3]-r1);         bl[2]=ddr2;
#pragma unroll
    for (int c = 0; c < 2; ++c) {
        Ap[(3+3*c)*NVa + c]      = 1.f;
        Ap[(3+3*c)*NVa + 12 + c] = -vv[2*c];
        Ap[(3+3*c)*NVa + 14 + c] = -vv[2*c+2];
        Ap[(4+3*c)*NVa + c + 2]  = 1.f;
        Ap[(4+3*c)*NVa + 12 + c] = -vv[2*c+1];
        Ap[(4+3*c)*NVa + 14 + c] = -vv[2*c+3];
        Ap[(5+3*c)*NVa + 12 + c] = 1.f;
        Ap[(5+3*c)*NVa + 14 + c] = 1.f;                     bl[5+3*c]=1.f;
        Ap[(9+2*c)*NVa + 16 + c] = fc[2*c];
        Ap[(9+2*c)*NVa + 18 + c] = fc[2*c+2];
        Ap[(9+2*c)*NVa + 4 + c]  = -1.f;
        Ap[(10+2*c)*NVa + 16 + c] = fc[2*c+1];
        Ap[(10+2*c)*NVa + 18 + c] = fc[2*c+3];
        Ap[(10+2*c)*NVa + 6 + c]  = -1.f;
    }
    Ap[13*NVa+20]=fce[0]; Ap[13*NVa+21]=fce[2]; Ap[13*NVa+8]=-1.f;
    Ap[14*NVa+20]=fce[1]; Ap[14*NVa+21]=fce[3]; Ap[14*NVa+10]=-1.f;
    Ap[15*NVa+22]=fce[4]; Ap[15*NVa+23]=fce[6]; Ap[15*NVa+9]=-1.f;
    Ap[16*NVa+22]=fce[5]; Ap[16*NVa+23]=fce[7]; Ap[16*NVa+11]=-1.f;

#pragma unroll
    for (int e = 0; e < NEq; ++e) bvec[(size_t)p * NEq + e] = bl[e];
}

// ---------------------------------------------------------------------------
// Fused spectral norm (25 power iters) + PDHG (400 iters).
// One wave per LP problem, 8 problems per 256-thread block. A lives in LDS,
// staged via CDNA5 async global->LDS DMA (ASYNCcnt).
// ---------------------------------------------------------------------------
__global__ void __launch_bounds__(256)
solve_pdhg(const float* __restrict__ Ag, const float* __restrict__ bg,
           float* __restrict__ zg)
{
    __shared__ float As[8][NEq * AROW];
    __shared__ float zs[8][NVa];
    __shared__ float zbs[8][NVa];
    __shared__ float ys[8][NEq];
    __shared__ float us[8][NVa];
    __shared__ float wt[8][NVa];
    __shared__ float bc[8];

    const int s    = threadIdx.x >> 5;
    const int lane = threadIdx.x & 31;
    const int p    = blockIdx.x * 8 + s;

    // ---- async DMA A tile into LDS (global_load_async_to_lds_b32, GV mode) ----
    {
        const float* Apg = Ag + (size_t)p * (NEq * NVa);
        uint32_t ldsbase = (uint32_t)(uintptr_t)(&As[s][0]);  // low 32 bits = LDS offset
        for (int i = lane; i < NEq * NVa; i += 32) {
            int e = i / NVa, v = i - e * NVa;
            uint32_t laddr = ldsbase + (uint32_t)(e * AROW + v) * 4u;
            uint64_t gaddr = (uint64_t)(uintptr_t)(Apg + i);
            asm volatile("global_load_async_to_lds_b32 %0, %1, off"
                         :: "v"(laddr), "v"(gaddr) : "memory");
        }
    }
    if (lane < NVa) us[s][lane] = 1.0f;
    asm volatile("s_wait_asynccnt 0x0" ::: "memory");
    __syncthreads();

    // ---- power iteration: u = normalize(A^T (A u)) ----
    for (int it = 0; it < POW_ITERS; ++it) {
        if (lane < NEq) {
            float acc = 0.f;
#pragma unroll
            for (int v = 0; v < NVa; ++v) acc += As[s][lane * AROW + v] * us[s][v];
            ys[s][lane] = acc;
        }
        __syncthreads();
        if (lane < NVa) {
            float acc = 0.f;
#pragma unroll
            for (int e = 0; e < NEq; ++e) acc += As[s][e * AROW + lane] * ys[s][e];
            wt[s][lane] = acc;
        }
        __syncthreads();
        if (lane == 0) {
            float nn = 0.f;
            for (int v = 0; v < NVa; ++v) nn += wt[s][v] * wt[s][v];
            bc[s] = 1.0f / (sqrtf(nn) + 1e-12f);
        }
        __syncthreads();
        if (lane < NVa) us[s][lane] = wt[s][lane] * bc[s];
        __syncthreads();
    }
    // L = ||A u||, tau = sig = 0.9 / (L + 1e-8)
    if (lane < NEq) {
        float acc = 0.f;
#pragma unroll
        for (int v = 0; v < NVa; ++v) acc += As[s][lane * AROW + v] * us[s][v];
        ys[s][lane] = acc;
    }
    __syncthreads();
    if (lane == 0) {
        float nn = 0.f;
        for (int e = 0; e < NEq; ++e) nn += ys[s][e] * ys[s][e];
        bc[s] = 0.9f / (sqrtf(nn) + 1e-8f);
    }
    __syncthreads();
    const float tau = bc[s];

    float be = 0.f, yreg = 0.f;
    if (lane < NEq) be = bg[(size_t)p * NEq + lane];
    if (lane < NVa) { zs[s][lane] = 0.f; zbs[s][lane] = 0.f; }
    if (lane < NEq) ys[s][lane] = 0.f;
    __syncthreads();

    // ---- PDHG main loop ----
    for (int it = 0; it < PDHG_ITERS; ++it) {
        if (lane < NEq) {
            float acc = -be;
#pragma unroll
            for (int v = 0; v < NVa; ++v) acc += As[s][lane * AROW + v] * zbs[s][v];
            yreg += tau * acc;
            ys[s][lane] = yreg;
        }
        __syncthreads();
        if (lane < NVa) {
            float acc = 0.f;
#pragma unroll
            for (int e = 0; e < NEq; ++e) acc += As[s][e * AROW + lane] * ys[s][e];
            float g = zs[s][lane] - tau * acc;
            float zn;
            if (lane >= 4 && lane < 8) {              // L1 coords: soft-threshold
                float ag = fabsf(g) - tau;
                zn = (ag > 0.f) ? copysignf(ag, g) : 0.f;
            } else if (lane >= 12) {                  // nonneg coords
                zn = fmaxf(g, 0.f);
            } else {
                zn = g;
            }
            zbs[s][lane] = 2.f * zn - zs[s][lane];
            zs[s][lane]  = zn;
        }
        __syncthreads();
    }
    if (lane < NVa) zg[(size_t)p * NVa + lane] = zs[s][lane];
}

// ---------------------------------------------------------------------------
// Output assembly: out = 100 * [p, f, 10*dp, 10*dv]  -> (B,100)
// pr stride 32, v stride 48.
// ---------------------------------------------------------------------------
__global__ void __launch_bounds__(256)
finalize(const float* __restrict__ z, const float* __restrict__ pr,
         const float* __restrict__ v, const float* __restrict__ x,
         float* __restrict__ out)
{
    int idx = blockIdx.x * blockDim.x + threadIdx.x;
    if (idx >= BATCH * 100) return;
    int b = idx / 100, j = idx - b * 100;
    float r;
    if (j < 20) {
        int c = j / 5, t = j - c * 5;
        r = 100.f * z[(size_t)(b * TSTEP + t) * NVa + c];
    } else if (j < 40) {
        int jj = j - 20; int c = jj / 5, t = jj - c * 5;
        r = 100.f * z[(size_t)(b * TSTEP + t) * NVa + 4 + c];
    } else if (j < 60) {
        int jj = j - 40;
        r = 1000.f * (pr[(size_t)b * 32 + jj] - x[(size_t)b * 160 + jj]);
    } else {
        int jj = j - 60;
        r = 1000.f * (v[(size_t)b * 48 + jj] - x[(size_t)b * 160 + 120 + jj]);
    }
    out[idx] = r;
}

// ---------------------------------------------------------------------------
static inline int gemm_blocks(int M, int Nrun)
{
    int waves = (M / 16) * (Nrun / 16);
    return (waves + 3) / 4;   // 4 waves per 128-thread block
}

extern "C" void kernel_launch(void* const* d_in, const int* in_sizes, int n_in,
                              void* d_out, int out_size, void* d_ws, size_t ws_size,
                              hipStream_t stream)
{
    (void)in_sizes; (void)n_in; (void)out_size; (void)ws_size;
    const float* xtraj = (const float*)d_in[0];
    const float* x     = (const float*)d_in[1];
    const float* ximg  = (const float*)d_in[2];
    const float* cw1 = (const float*)d_in[3];  const float* cb1 = (const float*)d_in[4];
    const float* cw2 = (const float*)d_in[5];  const float* cb2 = (const float*)d_in[6];
    const float* w3  = (const float*)d_in[7];  const float* b3  = (const float*)d_in[8];
    const float* w4  = (const float*)d_in[9];  const float* b4  = (const float*)d_in[10];
    const float* pw1 = (const float*)d_in[11]; const float* pb1 = (const float*)d_in[12];
    const float* pw2 = (const float*)d_in[13]; const float* pb2 = (const float*)d_in[14];
    const float* pw3 = (const float*)d_in[15]; const float* pb3 = (const float*)d_in[16];
    const float* vw1 = (const float*)d_in[17]; const float* vb1 = (const float*)d_in[18];
    const float* vw2 = (const float*)d_in[19]; const float* vb2 = (const float*)d_in[20];
    const float* vw3 = (const float*)d_in[21]; const float* vb3 = (const float*)d_in[22];

    // workspace layout (all offsets kept 16-float = 64B aligned)
    float* ptr = (float*)d_ws;
    auto alloc = [&ptr](size_t n) { float* p = ptr; ptr += (n + 15) & ~(size_t)15; return p; };

    float* h1  = alloc(2048000);       // conv1 pooled (B,10,20,20)
    float* e2  = alloc(BATCH * 320);   // conv2 flat
    float* f3  = alloc(BATCH * 160);   // fc3 out (150 -> pad 160)
    float* e4  = alloc(BATCH * 112);   // fc4 out (100 -> pad 112)
    float* h   = alloc(BATCH * 160);   // concat (145 -> pad 160)
    float* m1  = alloc(BATCH * 128);   // p-head L1 (100 -> pad 128)
    float* m2  = alloc(BATCH * 128);
    float* prb = alloc(BATCH * 32);    // p_r (20 -> pad 32)
    float* n1  = alloc(BATCH * 128);   // v-head L1
    float* n2  = alloc(BATCH * 128);
    float* vbf = alloc(BATCH * 48);    // v (40 -> pad 48)
    float* Ab  = alloc((size_t)NPROB * NEq * NVa);
    float* bb  = alloc((size_t)NPROB * NEq);
    float* zb  = alloc((size_t)NPROB * NVa);
    // padded f16 weights + f32 biases
    _Float16* w3p  = (_Float16*)alloc((160 * 320) / 2);
    _Float16* w4p  = (_Float16*)alloc((112 * 160) / 2);
    _Float16* pw1p = (_Float16*)alloc((128 * 160) / 2);
    _Float16* pw2p = (_Float16*)alloc((128 * 128) / 2);
    _Float16* pw3p = (_Float16*)alloc((32 * 128) / 2);
    _Float16* vw1p = (_Float16*)alloc((128 * 160) / 2);
    _Float16* vw2p = (_Float16*)alloc((128 * 128) / 2);
    _Float16* vw3p = (_Float16*)alloc((48 * 128) / 2);
    float* b3p  = alloc(160);
    float* b4p  = alloc(112);
    float* pb1p = alloc(128);
    float* pb2p = alloc(128);
    float* pb3p = alloc(32);
    float* vb1p = alloc(128);
    float* vb2p = alloc(128);
    float* vb3p = alloc(48);

    // ---- pack weights/biases into padded f16/f32 ----
#define PACKW(Wsrc, Wdst, N, K, NR, KP) \
    pack_w<<<((NR)*(KP) + 255) / 256, 256, 0, stream>>>(Wsrc, Wdst, N, K, NR, KP)
#define PACKB(Bsrc, Bdst, N, NR) \
    pack_b<<<((NR) + 255) / 256, 256, 0, stream>>>(Bsrc, Bdst, N, NR)
    PACKW(w3,  w3p,  150, 320, 160, 320);  PACKB(b3,  b3p,  150, 160);
    PACKW(w4,  w4p,  100, 150, 112, 160);  PACKB(b4,  b4p,  100, 112);
    PACKW(pw1, pw1p, 100, 145, 128, 160);  PACKB(pb1, pb1p, 100, 128);
    PACKW(pw2, pw2p, 100, 100, 128, 128);  PACKB(pb2, pb2p, 100, 128);
    PACKW(pw3, pw3p,  20, 100,  32, 128);  PACKB(pb3, pb3p,  20,  32);
    PACKW(vw1, vw1p, 100, 145, 128, 160);  PACKB(vb1, vb1p, 100, 128);
    PACKW(vw2, vw2p, 100, 100, 128, 128);  PACKB(vb2, vb2p, 100, 128);
    PACKW(vw3, vw3p,  40, 100,  48, 128);  PACKB(vb3, vb3p,  40,  48);
#undef PACKW
#undef PACKB

    // ---- encoder ----
    conv1_pool<<<(BATCH * 10 * 400 + 255) / 256, 256, 0, stream>>>(ximg, cw1, cb1, h1);
    conv2_pool<<<(BATCH * 20 * 16 + 255) / 256, 256, 0, stream>>>(h1, cw2, cb2, e2);
    wmma_gemm<<<gemm_blocks(BATCH, 160), 128, 0, stream>>>(e2, w3p, b3p, f3, BATCH, 160, 320, 1);
    wmma_gemm<<<gemm_blocks(BATCH, 112), 128, 0, stream>>>(f3, w4p, b4p, e4, BATCH, 112, 160, 1);
    concat_h<<<(BATCH * 160 + 255) / 256, 256, 0, stream>>>(e4, xtraj, h);
    // ---- p_r head ----
    wmma_gemm<<<gemm_blocks(BATCH, 128), 128, 0, stream>>>(h,  pw1p, pb1p, m1, BATCH, 128, 160, 1);
    wmma_gemm<<<gemm_blocks(BATCH, 128), 128, 0, stream>>>(m1, pw2p, pb2p, m2, BATCH, 128, 128, 1);
    wmma_gemm<<<gemm_blocks(BATCH, 32),  128, 0, stream>>>(m2, pw3p, pb3p, prb, BATCH, 32, 128, 0);
    // ---- v head ----
    wmma_gemm<<<gemm_blocks(BATCH, 128), 128, 0, stream>>>(h,  vw1p, vb1p, n1, BATCH, 128, 160, 1);
    wmma_gemm<<<gemm_blocks(BATCH, 128), 128, 0, stream>>>(n1, vw2p, vb2p, n2, BATCH, 128, 128, 1);
    wmma_gemm<<<gemm_blocks(BATCH, 48),  128, 0, stream>>>(n2, vw3p, vb3p, vbf, BATCH, 48, 128, 0);
    // ---- LP build + solve ----
    build_Ab<<<(NPROB + 255) / 256, 256, 0, stream>>>(xtraj, x, vbf, prb, Ab, bb);
    solve_pdhg<<<NPROB / 8, 256, 0, stream>>>(Ab, bb, zb);
    // ---- output ----
    finalize<<<(BATCH * 100 + 255) / 256, 256, 0, stream>>>(zb, prb, vbf, x, (float*)d_out);
}
